// DimeNetPPEnergy_Dropout_944892805682
// MI455X (gfx1250) — compile-verified, matrix operand order checked by reading, not run on hardware
//
#include <hip/hip_runtime.h>

typedef unsigned short u16;
typedef __attribute__((ext_vector_type(16))) __bf16 v16bf;
typedef __attribute__((ext_vector_type(8)))  float  v8f;

#define GF_SWISH 1
#define GF_MUL   2
#define GF_ADD   4

// ---------- bf16 helpers ----------
__device__ __forceinline__ float bf2f(u16 u) {
  union { unsigned int i; float f; } x; x.i = ((unsigned int)u) << 16; return x.f;
}
__device__ __forceinline__ u16 f2bf(float f) {
  union { float f; unsigned int i; } x; x.f = f;
  unsigned int r = x.i + 0x7fffu + ((x.i >> 16) & 1u);
  return (u16)(r >> 16);
}

// ---------- WMMA GEMM: Y[M,N](bf16) = epilogue( X[M,KK](bf16) @ WT[N,KK]^T(bf16) ) ----------
// WT (row-major transpose of the weight) is staged into LDS once per block (shared by all
// 8 waves), rows padded by 8 halves so B-fragment ds_load_b128 is bank-conflict-free.
// NT and KK are compile-time: accumulators stay in VGPRs, all offsets are immediates.
template <int NT, int KK>
__global__ __launch_bounds__(256) void gemm_bf16_kernel(
    const u16* __restrict__ X, const u16* __restrict__ WT,
    const float* __restrict__ bias, const u16* __restrict__ mulb,
    const u16* __restrict__ resb, u16* __restrict__ Y,
    int M, int N, int flags)
{
  constexpr int COLS = NT * 16;
  constexpr int KP = KK + 8;                    // +16B pad per row: banks spread 4*col
  __shared__ __align__(16) u16 ldsb[COLS * KP];

  const int wave = threadIdx.x >> 5;
  const int lane = threadIdx.x & 31;
  const int lr   = lane & 15;
  const int khalf = (lane >> 4) << 3;           // lanes 16-31 hold K +8 / +24 halves
  const int rowBase = blockIdx.x * 128 + wave * 16;
  const int n0 = blockIdx.y * COLS;

  // ---- cooperative WT slab stage: [COLS x KK] -> LDS (padded) ----
  {
    const u16* src = WT + (unsigned)(n0 * KK);
    for (int h8 = threadIdx.x; h8 < COLS * KK / 8; h8 += 256) {
      const int h = h8 * 8;
      const int row = h / KK, k = h - row * KK;   // KK multiple of 8: no row straddle
      *(uint4*)&ldsb[row * KP + k] = *(const uint4*)(src + h);
    }
  }
  __syncthreads();

  int arow = rowBase + lr; if (arow >= M) arow = M - 1;
  const u16* aptr = X + (unsigned)(arow * KK) + khalf;

  int boff[NT];
#pragma unroll
  for (int nt = 0; nt < NT; ++nt) boff[nt] = (nt * 16 + lr) * KP + khalf;

  v8f acc[NT];
#pragma unroll
  for (int nt = 0; nt < NT; ++nt)
    acc[nt] = (v8f){0.f,0.f,0.f,0.f,0.f,0.f,0.f,0.f};

#pragma unroll
  for (int k0 = 0; k0 < KK; k0 += 32) {
    union { v16bf v; uint4 q[2]; } a;
    a.q[0] = *(const uint4*)(aptr + k0);
    a.q[1] = *(const uint4*)(aptr + k0 + 16);
#pragma unroll
    for (int nt = 0; nt < NT; ++nt) {
      union { v16bf v; uint4 q[2]; } b;
      b.q[0] = *(const uint4*)&ldsb[boff[nt] + k0];
      b.q[1] = *(const uint4*)&ldsb[boff[nt] + k0 + 16];
      acc[nt] = __builtin_amdgcn_wmma_f32_16x16x32_bf16(
          false, a.v, false, b.v, (short)0, acc[nt], false, false);
    }
  }

  // C/D layout: VGPR r -> row rowBase+r (lanes 0-15) / rowBase+8+r (lanes 16-31),
  // col n0+nt*16+(lane&15)
  const int rowOff = (lane >> 4) << 3;
#pragma unroll
  for (int nt = 0; nt < NT; ++nt) {
    const int col = n0 + nt * 16 + lr;
    const float bv = bias ? bias[col] : 0.f;
#pragma unroll
    for (int r = 0; r < 8; ++r) {
      const int row = rowBase + rowOff + r;
      if (row < M) {
        float y = acc[nt][r] + bv;
        if (flags & GF_SWISH) y = y * (1.f / (1.f + __expf(-y)));
        const size_t o = (size_t)row * N + col;
        if (flags & GF_MUL) y *= bf2f(mulb[o]);
        if (flags & GF_ADD) y += bf2f(resb[o]);
        Y[o] = f2bf(y);
      }
    }
  }
}

// ---------- weight prep: W[K,N] f32 -> WT[N,K] bf16 ----------
__global__ void transpose_w_kernel(const float* __restrict__ W, u16* __restrict__ WT,
                                   int K, int N) {
  int i = blockIdx.x * 256 + threadIdx.x;
  if (i >= N * K) return;
  int n = i / K, k = i - n * K;
  WT[i] = f2bf(W[(size_t)k * N + n]);
}

// ---------- tiny chain collapse: C[J,Cn] = A[J,8] @ B[8,Cn] ----------
__global__ void combine8_kernel(const float* __restrict__ A, const float* __restrict__ B,
                                float* __restrict__ C, int J, int Cn) {
  int i = blockIdx.x * 256 + threadIdx.x;
  if (i >= J * Cn) return;
  int j = i / Cn, c = i - j * Cn;
  float s = 0.f;
#pragma unroll
  for (int t = 0; t < 8; ++t) s += A[j * 8 + t] * B[t * Cn + c];
  C[i] = s;
}

// ---------- embedding concat: [h_i(64) | h_j(64) | rbf@Wemb(128)] -> bf16 [E,256] ----------
__global__ void edge_concat_kernel(const float* __restrict__ embed, const float* __restrict__ Wemb,
                                   const float* __restrict__ rbf, const int* __restrict__ species,
                                   const int* __restrict__ idx_i, const int* __restrict__ idx_j,
                                   u16* __restrict__ out, int E) {
  int i = blockIdx.x * 256 + threadIdx.x;
  if (i >= E * 256) return;
  int e = i >> 8, t = i & 255;
  float v;
  if (t < 64)        v = embed[species[idx_i[e]] * 64 + t];
  else if (t < 128)  v = embed[species[idx_j[e]] * 64 + (t - 64)];
  else {
    int c = t - 128; v = 0.f;
    const float* rr = rbf + (size_t)e * 6;
#pragma unroll
    for (int j = 0; j < 6; ++j) v += rr[j] * Wemb[j * 128 + c];
  }
  out[i] = f2bf(v);
}

// ---------- r = rbf @ C6 -> bf16 [E,128] ----------
__global__ void rbf_proj_kernel(const float* __restrict__ rbf, const float* __restrict__ C6,
                                u16* __restrict__ out, int E) {
  int i = blockIdx.x * 256 + threadIdx.x;
  if (i >= E * 128) return;
  int e = i >> 7, c = i & 127;
  const float* rr = rbf + (size_t)e * 6;
  float v = 0.f;
#pragma unroll
  for (int j = 0; j < 6; ++j) v += rr[j] * C6[j * 128 + c];
  out[i] = f2bf(v);
}

// ---------- angle stage: agg[reduce[a],c] += m_ang[expand[a],c] * (sbf[a] . sbfC[:,c]) ----------
__global__ __launch_bounds__(256) void angle_kernel(
    const float* __restrict__ sbf, const float* __restrict__ sbfC,
    const u16* __restrict__ mang, const int* __restrict__ expand,
    const int* __restrict__ reduce, float* __restrict__ agg, int A) {
  __shared__ float sC[42 * 64];
  __shared__ float srow[4 * 42];
  for (int i = threadIdx.x; i < 42 * 64; i += 256) sC[i] = sbfC[i];
  const int a0 = blockIdx.x * 4;
  for (int i = threadIdx.x; i < 4 * 42; i += 256) {
    int aa = a0 + i / 42;
    srow[i] = (aa < A) ? sbf[(size_t)aa * 42 + (i % 42)] : 0.f;
  }
  __syncthreads();
  const int al = threadIdx.x >> 6, c = threadIdx.x & 63;
  const int a = a0 + al;
  if (a >= A) return;
  const float* rr = srow + al * 42;
  float s = 0.f;
#pragma unroll
  for (int j = 0; j < 42; ++j) s += rr[j] * sC[j * 64 + c];
  float val = bf2f(mang[(size_t)expand[a] * 64 + c]) * s;
  unsafeAtomicAdd(&agg[(size_t)reduce[a] * 64 + c], val);
}

// ---------- f32 -> bf16 ----------
__global__ void cvt_bf16_kernel(const float* __restrict__ src, u16* __restrict__ dst, int n) {
  int i = blockIdx.x * 256 + threadIdx.x;
  if (i < n) dst[i] = f2bf(src[i]);
}

// ---------- output head front: seg[idx_i[e],c] += m[e,c] * (rbf[e] . C6[:,c]) ----------
__global__ void out_msg_kernel(const float* __restrict__ rbf, const float* __restrict__ C6,
                               const u16* __restrict__ m, const int* __restrict__ idx_i,
                               float* __restrict__ seg, int E) {
  int i = blockIdx.x * 256 + threadIdx.x;
  if (i >= E * 128) return;
  int e = i >> 7, c = i & 127;
  const float* rr = rbf + (size_t)e * 6;
  float t = 0.f;
#pragma unroll
  for (int j = 0; j < 6; ++j) t += rr[j] * C6[j * 128 + c];
  unsafeAtomicAdd(&seg[(size_t)idx_i[e] * 128 + c], bf2f(m[i]) * t);
}

// ---------- output head tail: total += sum_atom (u[atom,:] . wfin) ----------
__global__ __launch_bounds__(256) void out_final_kernel(const u16* __restrict__ u,
                                                        const float* __restrict__ wfin,
                                                        float* __restrict__ out, int A) {
  __shared__ float red[256];
  const int atom = blockIdx.x;
  float v = bf2f(u[(size_t)atom * 256 + threadIdx.x]) * wfin[threadIdx.x];
  red[threadIdx.x] = v;
  __syncthreads();
  for (int s = 128; s > 0; s >>= 1) {
    if ((int)threadIdx.x < s) red[threadIdx.x] += red[threadIdx.x + s];
    __syncthreads();
  }
  if (threadIdx.x == 0) unsafeAtomicAdd(out, red[0]);
}

// =====================================================================================

extern "C" void kernel_launch(void* const* d_in, const int* in_sizes, int n_in,
                              void* d_out, int out_size, void* d_ws, size_t ws_size,
                              hipStream_t stream) {
  (void)in_sizes; (void)n_in; (void)out_size; (void)ws_size;
  constexpr int E = 200000, NA = 800000, NATOM = 12500;
  constexpr int EMB = 128, ANG = 64, OUT = 256;

  // ---- parameter walk: setup_inputs() insertion order, nested depth-first, lin = (w[,b]) ----
  int pi = 0;
  auto nf = [&]() { return (const float*)d_in[pi++]; };
  auto ni = [&]() { return (const int*)d_in[pi++]; };
  struct Lin { const float *w, *b; };

  const float* embed_vect   = nf();
  const float* emb_rbf_w    = nf();
  Lin emb_concat            = { nf(), nf() };

  struct Blk {
    Lin kj; const float *rbf1, *rbf2, *sbf1, *sbf2, *down, *up; Lin ji;
    Lin rb0, rb1, fin; Lin ra[6];
  } blk[2];
  for (int i = 0; i < 2; ++i) {
    blk[i].kj   = { nf(), nf() };
    blk[i].rbf1 = nf(); blk[i].rbf2 = nf();
    blk[i].sbf1 = nf(); blk[i].sbf2 = nf();
    blk[i].down = nf(); blk[i].up   = nf();
    blk[i].ji   = { nf(), nf() };
    blk[i].rb0  = { nf(), nf() };
    blk[i].rb1  = { nf(), nf() };
    blk[i].fin  = { nf(), nf() };
    for (int j = 0; j < 6; ++j) blk[i].ra[j] = { nf(), nf() };
  }
  struct OutB { const float* rbfd; const float* up; Lin d[3]; const float* fin; } ob[3];
  for (int i = 0; i < 3; ++i) {
    ob[i].rbfd = nf(); ob[i].up = nf();
    for (int j = 0; j < 3; ++j) ob[i].d[j] = { nf(), nf() };
    ob[i].fin = nf();
  }
  const float* rbf = nf();
  const float* sbf = nf();
  const int* species      = ni();
  const int* idx_i        = ni();
  const int* idx_j        = ni();
  const int* reduce_to_ji = ni();
  const int* expand_to_kj = ni();

  // ---- workspace bump allocator ----
  char* wp = (char*)d_ws;
  auto alloc = [&](size_t b) -> void* {
    void* p = (void*)wp; wp += (b + 255) & ~(size_t)255; return p;
  };

  u16* concat = (u16*)alloc((size_t)E * 256 * 2);       // dead after embedding GEMM
  u16* m      = (u16*)alloc((size_t)E * EMB * 2);
  u16* t1     = (u16*)alloc((size_t)E * EMB * 2);
  u16* t2     = (u16*)alloc((size_t)E * EMB * 2);
  u16* rbuf   = (u16*)alloc((size_t)E * EMB * 2);       // r multiplier, then residual scratch
  // alias angle-stage buffers into the (then-dead) concat region: 25.6+51.2+25.6 = 102.4MB
  u16*   mang = (u16*)concat;                                       // [E,64] bf16
  float* agg  = (float*)((char*)concat + (size_t)E * ANG * 2);      // [E,64] f32
  u16*   aggb = (u16*)((char*)agg + (size_t)E * ANG * 4);           // [E,64] bf16
  float* seg  = (float*)alloc((size_t)NATOM * EMB * 4);
  u16*   segb = (u16*)alloc((size_t)NATOM * EMB * 2);
  u16*   u0   = (u16*)alloc((size_t)NATOM * OUT * 2);
  u16*   u1   = (u16*)alloc((size_t)NATOM * OUT * 2);

  auto prepW = [&](const float* W, int K, int N) -> u16* {
    u16* wt = (u16*)alloc((size_t)K * N * 2);
    int total = K * N;
    transpose_w_kernel<<<(total + 255) / 256, 256, 0, stream>>>(W, wt, K, N);
    return wt;
  };
  auto prepC8 = [&](const float* A, const float* B, int J, int Cn) -> float* {
    float* c = (float*)alloc((size_t)J * Cn * 4);
    int total = J * Cn;
    combine8_kernel<<<(total + 255) / 256, 256, 0, stream>>>(A, B, c, J, Cn);
    return c;
  };

  // ---- convert / collapse all weights once per launch (deterministic) ----
  u16* WTembcat = prepW(emb_concat.w, 256, 128);
  struct BlkW { u16 *kj, *down, *up, *ji, *rb0, *rb1, *fin, *ra[6]; float *rbfC, *sbfC; } bw[2];
  for (int i = 0; i < 2; ++i) {
    bw[i].kj   = prepW(blk[i].kj.w, 128, 128);
    bw[i].down = prepW(blk[i].down, 128, 64);
    bw[i].up   = prepW(blk[i].up, 64, 128);
    bw[i].ji   = prepW(blk[i].ji.w, 128, 128);
    bw[i].rb0  = prepW(blk[i].rb0.w, 128, 128);
    bw[i].rb1  = prepW(blk[i].rb1.w, 128, 128);
    bw[i].fin  = prepW(blk[i].fin.w, 128, 128);
    for (int j = 0; j < 6; ++j) bw[i].ra[j] = prepW(blk[i].ra[j].w, 128, 128);
    bw[i].rbfC = prepC8(blk[i].rbf1, blk[i].rbf2, 6, 128);
    bw[i].sbfC = prepC8(blk[i].sbf1, blk[i].sbf2, 42, 64);
  }
  struct OutW { u16 *up, *d[3]; } ow[3];
  for (int i = 0; i < 3; ++i) {
    ow[i].up = prepW(ob[i].up, 128, 256);
    for (int j = 0; j < 3; ++j) ow[i].d[j] = prepW(ob[i].d[j].w, 256, 256);
  }
  const float* outRbfW[3] = { ob[0].rbfd, ob[1].rbfd, ob[2].rbfd };

  auto gemm = [&](const u16* X, const u16* WT, const float* bias, const u16* mulb,
                  const u16* resb, u16* Y, int M, int K, int N, int flags) {
    dim3 g((M + 127) / 128, (N + 127) / 128);
    if (N == 64) {
      gemm_bf16_kernel<4, 128><<<g, 256, 0, stream>>>(X, WT, bias, mulb, resb, Y, M, N, flags);
    } else if (K == 64) {
      gemm_bf16_kernel<8, 64><<<g, 256, 0, stream>>>(X, WT, bias, mulb, resb, Y, M, N, flags);
    } else if (K == 128) {
      gemm_bf16_kernel<8, 128><<<g, 256, 0, stream>>>(X, WT, bias, mulb, resb, Y, M, N, flags);
    } else {  // K == 256
      gemm_bf16_kernel<8, 256><<<g, 256, 0, stream>>>(X, WT, bias, mulb, resb, Y, M, N, flags);
    }
  };

  auto outHead = [&](int o, const u16* mcur) {
    hipMemsetAsync(seg, 0, (size_t)NATOM * EMB * 4, stream);
    out_msg_kernel<<<(E * 128 + 255) / 256, 256, 0, stream>>>(rbf, outRbfW[o], mcur, idx_i, seg, E);
    cvt_bf16_kernel<<<(NATOM * EMB + 255) / 256, 256, 0, stream>>>(seg, segb, NATOM * EMB);
    gemm(segb, ow[o].up, nullptr, nullptr, nullptr, u0, NATOM, 128, 256, 0);
    gemm(u0, ow[o].d[0], ob[o].d[0].b, nullptr, nullptr, u1, NATOM, 256, 256, GF_SWISH);
    gemm(u1, ow[o].d[1], ob[o].d[1].b, nullptr, nullptr, u0, NATOM, 256, 256, GF_SWISH);
    gemm(u0, ow[o].d[2], ob[o].d[2].b, nullptr, nullptr, u1, NATOM, 256, 256, GF_SWISH);
    out_final_kernel<<<NATOM, 256, 0, stream>>>(u1, ob[o].fin, (float*)d_out, NATOM);
  };

  // ---- forward ----
  hipMemsetAsync(d_out, 0, sizeof(float), stream);

  // embedding: concat -> swish(lin 256->128)
  edge_concat_kernel<<<(E * 256 + 255) / 256, 256, 0, stream>>>(
      embed_vect, emb_rbf_w, rbf, species, idx_i, idx_j, concat, E);
  gemm(concat, WTembcat, emb_concat.b, nullptr, nullptr, m, E, 256, 128, GF_SWISH);

  outHead(0, m);

  for (int b = 0; b < 2; ++b) {
    // r multiplier, then m_ang = swish(dense_kj(m)) * r
    rbf_proj_kernel<<<(E * 128 + 255) / 256, 256, 0, stream>>>(rbf, bw[b].rbfC, rbuf, E);
    gemm(m, bw[b].kj, blk[b].kj.b, rbuf, nullptr, t1, E, 128, 128, GF_SWISH | GF_MUL);
    // down-project to ANG
    gemm(t1, bw[b].down, nullptr, nullptr, nullptr, mang, E, 128, 64, GF_SWISH);
    // angle gather * sbf-proj, f32 segment sum
    hipMemsetAsync(agg, 0, (size_t)E * ANG * 4, stream);
    angle_kernel<<<(NA + 3) / 4, 256, 0, stream>>>(sbf, bw[b].sbfC, mang, expand_to_kj,
                                                   reduce_to_ji, agg, NA);
    cvt_bf16_kernel<<<(E * ANG + 255) / 256, 256, 0, stream>>>(agg, aggb, E * ANG);
    // prop = swish(agg @ up)   (reuse t1)
    gemm(aggb, bw[b].up, nullptr, nullptr, nullptr, t1, E, 64, 128, GF_SWISH);
    // mc = swish(dense_ji(m)) + prop
    gemm(m, bw[b].ji, blk[b].ji.b, nullptr, t1, t2, E, 128, 128, GF_SWISH | GF_ADD);
    // res_before (1 residual)
    gemm(t2, bw[b].rb0, blk[b].rb0.b, nullptr, nullptr, rbuf, E, 128, 128, GF_SWISH);
    gemm(rbuf, bw[b].rb1, blk[b].rb1.b, nullptr, t2, t2, E, 128, 128, GF_SWISH | GF_ADD);
    // m_skip = swish(final(mc)) + m
    gemm(t2, bw[b].fin, blk[b].fin.b, nullptr, m, t1, E, 128, 128, GF_SWISH | GF_ADD);
    // res_after ×3 in-place on t1
    for (int j = 0; j < 3; ++j) {
      gemm(t1, bw[b].ra[2 * j], blk[b].ra[2 * j].b, nullptr, nullptr, rbuf, E, 128, 128, GF_SWISH);
      gemm(rbuf, bw[b].ra[2 * j + 1], blk[b].ra[2 * j + 1].b, nullptr, t1, t1, E, 128, 128,
           GF_SWISH | GF_ADD);
    }
    { u16* tmp = m; m = t1; t1 = tmp; }
    outHead(b + 1, m);
  }
}
